// SSAMLayer_86492051407235
// MI455X (gfx1250) — compile-verified, matrix operand order checked by reading, not run on hardware
//
#include <hip/hip_runtime.h>
#include <math.h>

// ---------------------------------------------------------------------------
// Problem constants (from reference): B=2, C=64, H=W=64, O=64, bands=4, GN=8
// ---------------------------------------------------------------------------
#define CCH 64
#define PX  4096          // H*W
#define NBB 8             // B * NUM_BANDS
#define BN_EPS 1e-5f

typedef __attribute__((ext_vector_type(16))) __bf16         v16bf;
typedef __attribute__((ext_vector_type(8)))  float          v8f;
typedef __attribute__((ext_vector_type(16))) unsigned short v16u;
typedef __attribute__((ext_vector_type(4)))  unsigned int   v4u;

union Frag {
  v16bf bf;
  v16u  us;
  v4u   q[2];
};

__device__ __forceinline__ unsigned short f2bf(float f) {
  union { float f; unsigned u; } v; v.f = f;
  unsigned r = v.u + 0x7FFFu + ((v.u >> 16) & 1u);   // round-to-nearest-even
  return (unsigned short)(r >> 16);
}

// A fragment (16x32 bf16 tile of row-major A[.., ldk]):
// lanes 0-15: row=lane, K = kbase+[0,8) and kbase+[16,24)
// lanes 16-31: row=lane-16, K = kbase+[8,16) and kbase+[24,32)
__device__ __forceinline__ v16bf load_a_frag(const unsigned short* A, int ldk,
                                             int row, int kbase, int lane) {
  int hi = lane >> 4;
  const unsigned short* p = A + (size_t)row * ldk + kbase + hi * 8;
  Frag r;
  r.q[0] = *(const v4u*)p;
  r.q[1] = *(const v4u*)(p + 16);
  return r.bf;
}

// B fragment (32x16, K x N) from fp32 source with leading dim ld (K stride),
// converting to bf16.  lane holds column (lane&15); K runs mirror A's runs.
__device__ __forceinline__ v16bf load_b_frag_f32(const float* B, int ld,
                                                 int col, int kbase, int lane) {
  int hi = lane >> 4;
  const float* p = B + (size_t)(kbase + hi * 8) * ld + col;
  Frag r;
  #pragma unroll
  for (int i = 0; i < 8; ++i) r.us[i] = f2bf(p[(size_t)i * ld]);
  p += (size_t)16 * ld;
  #pragma unroll
  for (int i = 0; i < 8; ++i) r.us[8 + i] = f2bf(p[(size_t)i * ld]);
  return r.bf;
}

// ---------------------------------------------------------------------------
// Prep kernels
// ---------------------------------------------------------------------------

// Band-projection matrices: M[band][c][c'] = (1/64) sum_{f in band} w_f cos(2pi f (c-c')/64)
__global__ void prep_mband(unsigned short* __restrict__ Mbf) {
  int idx = blockIdx.x * blockDim.x + threadIdx.x;
  if (idx >= 4 * 64 * 64) return;
  int band = idx >> 12;
  int ci = (idx >> 6) & 63;
  int cj = idx & 63;
  int f0 = band * 8;
  int f1 = (band == 3) ? 33 : (band + 1) * 8;
  float d = (float)(ci - cj);
  float s = 0.0f;
  for (int f = f0; f < f1; ++f) {
    float w = (f == 0 || f == 32) ? 1.0f : 2.0f;
    s += w * cosf(6.28318530717958647692f * (float)f * d * (1.0f / 64.0f));
  }
  Mbf[idx] = f2bf(s * (1.0f / 64.0f));
}

// wbf[band][o][c*8+g] = bf16(spline_w * scalers)   (same flat order as inputs)
__global__ void prep_wspline(const float* __restrict__ sw, const float* __restrict__ sc,
                             unsigned short* __restrict__ wbf) {
  int idx = blockIdx.x * blockDim.x + threadIdx.x;
  if (idx >= 4 * 64 * 64 * 8) return;
  wbf[idx] = f2bf(sw[idx] * sc[idx >> 3]);
}

// pw weights -> bf16; fold BN params into scale/bias
__global__ void prep_pw_bn(const float* __restrict__ pw, unsigned short* __restrict__ pwbf,
                           const float* __restrict__ g1, const float* __restrict__ b1,
                           const float* __restrict__ m1, const float* __restrict__ v1,
                           const float* __restrict__ g2, const float* __restrict__ b2,
                           const float* __restrict__ m2, const float* __restrict__ v2,
                           float* __restrict__ sc1, float* __restrict__ bi1,
                           float* __restrict__ sc2, float* __restrict__ bi2) {
  int idx = blockIdx.x * blockDim.x + threadIdx.x;
  if (idx < 4096) pwbf[idx] = f2bf(pw[idx]);
  if (idx < 64) {
    float s1 = g1[idx] * rsqrtf(v1[idx] + BN_EPS);
    sc1[idx] = s1;
    bi1[idx] = b1[idx] - m1[idx] * s1;
    float s2 = g2[idx] * rsqrtf(v2[idx] + BN_EPS);
    sc2[idx] = s2;
    bi2[idx] = b2[idx] - m2[idx] * s2;
  }
}

// ---------------------------------------------------------------------------
// Generic 64xK(=64) GEMM via WMMA:  C[bb] = opt_bn( A[bb] @ B[bb>>b_shift] )
// A is bf16 [64][64] (optionally per-band via a_band_sel), B/C fp32 [64][4096].
// One wave computes a 64(row) x 16(pixel) tile; block = 8 waves = 128 pixels.
// ---------------------------------------------------------------------------
__global__ void __launch_bounds__(256)
gemm64_wmma(const unsigned short* __restrict__ Abase, int a_band_sel, int b_shift,
            const float* __restrict__ Bsrc, float* __restrict__ Csrc,
            const float* __restrict__ scale, const float* __restrict__ bias) {
  int lane = threadIdx.x & 31;
  int wave = threadIdx.x >> 5;
  int bb = blockIdx.y;
  int p0 = blockIdx.x * 128 + wave * 16;
  const unsigned short* A = Abase + (a_band_sel ? (size_t)(bb & 3) * 4096 : (size_t)0);
  const float* Bp = Bsrc + (size_t)(bb >> b_shift) * CCH * PX;
  float* Cp = Csrc + (size_t)bb * CCH * PX;

  v8f acc[4] = {};
  #pragma unroll
  for (int ks = 0; ks < 2; ++ks) {
    v16bf bfrag = load_b_frag_f32(Bp, PX, p0 + (lane & 15), ks * 32, lane);
    #pragma unroll
    for (int mt = 0; mt < 4; ++mt) {
      v16bf afrag = load_a_frag(A, 64, mt * 16 + (lane & 15), ks * 32, lane);
      acc[mt] = __builtin_amdgcn_wmma_f32_16x16x32_bf16(false, afrag, false, bfrag,
                                                        (short)0, acc[mt], false, false);
    }
  }
  int n = lane & 15, hi = lane >> 4;
  #pragma unroll
  for (int mt = 0; mt < 4; ++mt) {
    #pragma unroll
    for (int r = 0; r < 8; ++r) {
      int o = mt * 16 + r + hi * 8;
      float v = acc[mt][r];
      if (scale) v = v * scale[o] + bias[o];
      Cp[(size_t)o * PX + p0 + n] = v;
    }
  }
}

// ---------------------------------------------------------------------------
// Depthwise 3x3 conv + BN1 + ReLU (zero pad)
// ---------------------------------------------------------------------------
__global__ void dwconv_bn_relu(const float* __restrict__ bands, const float* __restrict__ dw,
                               const float* __restrict__ sc1, const float* __restrict__ bi1,
                               float* __restrict__ y1) {
  int idx = blockIdx.x * blockDim.x + threadIdx.x;   // (bb*64+c)*4096 + p
  int p = idx & 4095;
  int c = (idx >> 12) & 63;
  int h = p >> 6, w = p & 63;
  const float* src = bands + (size_t)(idx >> 12) * PX;
  float s = 0.0f;
  #pragma unroll
  for (int i = 0; i < 3; ++i) {
    int hh = h + i - 1;
    if (hh < 0 || hh > 63) continue;
    #pragma unroll
    for (int j = 0; j < 3; ++j) {
      int ww = w + j - 1;
      if (ww < 0 || ww > 63) continue;
      s += dw[c * 9 + i * 3 + j] * src[hh * 64 + ww];
    }
  }
  float v = s * sc1[c] + bi1[c];
  y1[idx] = v > 0.0f ? v : 0.0f;
}

// ---------------------------------------------------------------------------
// Global average pool over pixels: gap[bb*64+c]
// ---------------------------------------------------------------------------
__global__ void gap_kernel(const float* __restrict__ y2, float* __restrict__ gap) {
  __shared__ float red[256];
  int bbc = blockIdx.x;                       // 0..511
  const float* y = y2 + (size_t)bbc * PX;
  float s = 0.0f;
  for (int p = threadIdx.x; p < PX; p += 256) s += y[p];
  red[threadIdx.x] = s;
  __syncthreads();
  for (int off = 128; off > 0; off >>= 1) {
    if ((int)threadIdx.x < off) red[threadIdx.x] += red[threadIdx.x + off];
    __syncthreads();
  }
  if (threadIdx.x == 0) gap[bbc] = red[0] * (1.0f / (float)PX);
}

// ---------------------------------------------------------------------------
// Channel attention: cw = sigmoid(W2 @ gelu(W1 @ gap)), hid = 8
// ---------------------------------------------------------------------------
__global__ void chan_att(const float* __restrict__ gap, const float* __restrict__ w1,
                         const float* __restrict__ w2, float* __restrict__ cw) {
  __shared__ float g[64];
  __shared__ float t[8];
  int bb = blockIdx.x;
  int tid = threadIdx.x;
  g[tid] = gap[bb * 64 + tid];
  __syncthreads();
  if (tid < 8) {
    float s = 0.0f;
    for (int c = 0; c < 64; ++c) s += w1[tid * 64 + c] * g[c];
    t[tid] = 0.5f * s * (1.0f + erff(s * 0.70710678118654752f));  // exact GELU
  }
  __syncthreads();
  float s = 0.0f;
  #pragma unroll
  for (int h = 0; h < 8; ++h) s += w2[tid * 8 + h] * t[h];
  cw[bb * 64 + tid] = 1.0f / (1.0f + expf(-s));
}

// ---------------------------------------------------------------------------
// Mean / max over channels of (y2 * cw)
// ---------------------------------------------------------------------------
__global__ void mean_max_c(const float* __restrict__ y2, const float* __restrict__ cw,
                           float* __restrict__ mp) {
  int idx = blockIdx.x * blockDim.x + threadIdx.x;  // bb*4096 + p
  int bb = idx >> 12;
  int p = idx & 4095;
  const float* y = y2 + (size_t)bb * CCH * PX + p;
  const float* w = cw + bb * 64;
  float sm = 0.0f, mx = -3.402823466e38f;
  for (int c = 0; c < 64; ++c) {
    float v = y[(size_t)c * PX] * w[c];
    sm += v;
    mx = v > mx ? v : mx;
  }
  mp[(size_t)(bb * 2) * PX + p] = sm * (1.0f / 64.0f);
  mp[(size_t)(bb * 2 + 1) * PX + p] = mx;
}

// ---------------------------------------------------------------------------
// Spatial attention: 3x3 conv over [mean,max] (2ch -> 1ch), sigmoid
// ---------------------------------------------------------------------------
__global__ void spatial_att(const float* __restrict__ mp, const float* __restrict__ sa_w,
                            float* __restrict__ s) {
  int idx = blockIdx.x * blockDim.x + threadIdx.x;  // bb*4096 + p
  int bb = idx >> 12, p = idx & 4095;
  int h = p >> 6, w = p & 63;
  float acc = 0.0f;
  #pragma unroll
  for (int ch = 0; ch < 2; ++ch) {
    const float* src = mp + (size_t)(bb * 2 + ch) * PX;
    #pragma unroll
    for (int i = 0; i < 3; ++i) {
      int hh = h + i - 1;
      if (hh < 0 || hh > 63) continue;
      #pragma unroll
      for (int j = 0; j < 3; ++j) {
        int ww = w + j - 1;
        if (ww < 0 || ww > 63) continue;
        acc += sa_w[ch * 9 + i * 3 + j] * src[hh * 64 + ww];
      }
    }
  }
  s[idx] = 1.0f / (1.0f + expf(-acc));
}

// xatt = y2 * cw * s
__global__ void apply_att(const float* __restrict__ y2, const float* __restrict__ cw,
                          const float* __restrict__ s, float* __restrict__ xatt) {
  int idx = blockIdx.x * blockDim.x + threadIdx.x;  // (bb*64+c)*4096 + p
  int p = idx & 4095;
  int cc = idx >> 12;        // bb*64+c
  int bb = cc >> 6;
  int c = cc & 63;
  xatt[idx] = y2[idx] * cw[bb * 64 + c] * s[bb * 4096 + p];
}

// ---------------------------------------------------------------------------
// Cubic B-spline basis (uniform grid, h=2/5, grid[i]=(i-3)h-1, i=0..11)
// Degree-0 indicators on 11 intervals, 3 Cox-de Boor passes -> 8 coefficients.
// Denominators are exactly k*h (uniform grid) so no max(.,1e-6) needed.
// ---------------------------------------------------------------------------
__device__ __forceinline__ void spline_basis8(float x, float* out) {
  const float h = 0.4f;
  float b[11];
  #pragma unroll
  for (int i = 0; i < 11; ++i) {
    float g0 = (float)(i - 3) * h - 1.0f;
    float g1 = (float)(i - 2) * h - 1.0f;
    b[i] = (x >= g0 && x < g1) ? 1.0f : 0.0f;
  }
  #pragma unroll
  for (int k = 1; k <= 3; ++k) {
    float inv = 1.0f / ((float)k * h);
    #pragma unroll
    for (int i = 0; i < 11 - k; ++i) {
      float gi   = (float)(i - 3) * h - 1.0f;       // grid[i]
      float gik1 = (float)(i - 2 + k) * h - 1.0f;   // grid[i+k+1]
      b[i] = (x - gi) * inv * b[i] + (gik1 - x) * inv * b[i + 1];
    }
  }
  #pragma unroll
  for (int g = 0; g < 8; ++g) out[g] = b[g];
}

// ---------------------------------------------------------------------------
// Spline contraction via WMMA.
// out[b][o][p] = 0.25 * sum_band sum_{c,g} basis(xatt[b,band,c,p])[g] * w[band,o,c*8+g]
// One wave: 16 pixels x 64 outputs; K = 512 (c*8+g) in 16 steps of 32;
// per k-step each lane computes 2 (pixel,channel) basis octets, packs bf16,
// ds_store_b128 into its wave's LDS tile, reads back B fragments as b128.
// ---------------------------------------------------------------------------
__global__ void __launch_bounds__(256)
spline_wmma(const float* __restrict__ xatt, const unsigned short* __restrict__ Wbf,
            float* __restrict__ out) {
  __shared__ __align__(16) unsigned short lds[8][16][32];  // [wave][pixel][klocal]
  int lane = threadIdx.x & 31;
  int wave = threadIdx.x >> 5;
  int b = blockIdx.y;
  int p0 = blockIdx.x * 128 + wave * 16;

  v8f acc[4] = {};
  for (int band = 0; band < 4; ++band) {
    const float* xa = xatt + (size_t)(b * 4 + band) * CCH * PX;
    const unsigned short* W = Wbf + (size_t)band * 64 * 512;
    for (int ks = 0; ks < 16; ++ks) {
      // stage 32x16 bf16 basis tile: K = c*8+g, c in [ks*4, ks*4+4)
      #pragma unroll
      for (int rpt = 0; rpt < 2; ++rpt) {
        int idx = lane + rpt * 32;       // 0..63 -> (c_local, pixel)
        int cl = idx >> 4;
        int n = idx & 15;
        float a = xa[(size_t)(ks * 4 + cl) * PX + p0 + n];
        float bs[8];
        spline_basis8(a, bs);
        v4u pk;
        #pragma unroll
        for (int j = 0; j < 4; ++j)
          pk[j] = (unsigned)f2bf(bs[2 * j]) | ((unsigned)f2bf(bs[2 * j + 1]) << 16);
        *(v4u*)&lds[wave][n][cl * 8] = pk;
      }
      __syncthreads();
      Frag bf;
      const unsigned short* lp = &lds[wave][lane & 15][(lane >> 4) * 8];
      bf.q[0] = *(const v4u*)lp;
      bf.q[1] = *(const v4u*)(lp + 16);
      #pragma unroll
      for (int mt = 0; mt < 4; ++mt) {
        v16bf afrag = load_a_frag(W, 512, mt * 16 + (lane & 15), ks * 32, lane);
        acc[mt] = __builtin_amdgcn_wmma_f32_16x16x32_bf16(false, afrag, false, bf.bf,
                                                          (short)0, acc[mt], false, false);
      }
      __syncthreads();
    }
  }
  int n = lane & 15, hi = lane >> 4;
  #pragma unroll
  for (int mt = 0; mt < 4; ++mt) {
    #pragma unroll
    for (int r = 0; r < 8; ++r) {
      int o = mt * 16 + r + hi * 8;
      out[((size_t)b * CCH + o) * PX + p0 + n] = 0.25f * acc[mt][r];
    }
  }
}

// ---------------------------------------------------------------------------
// Workspace layout (bytes)
// ---------------------------------------------------------------------------
static constexpr size_t OFF_MBF   = 0;          //  4*64*64 bf16      = 32768
static constexpr size_t OFF_WBF   = 32768;      //  4*64*512 bf16     = 262144
static constexpr size_t OFF_PWBF  = 294912;     //  64*64 bf16        = 8192
static constexpr size_t OFF_SC1   = 303104;     //  64 f32
static constexpr size_t OFF_BI1   = 303360;
static constexpr size_t OFF_SC2   = 303616;
static constexpr size_t OFF_BI2   = 303872;
static constexpr size_t OFF_GAP   = 304128;     //  512 f32
static constexpr size_t OFF_CW    = 306176;     //  512 f32
static constexpr size_t OFF_MP    = 308224;     //  8*2*4096 f32      = 262144
static constexpr size_t OFF_S     = 570368;     //  8*4096 f32        = 131072
static constexpr size_t OFF_BANDS = 1048576;    //  8*64*4096 f32     = 8 MB
static constexpr size_t OFF_Y1    = OFF_BANDS + 8388608;
static constexpr size_t OFF_Y2    = OFF_Y1 + 8388608;
static constexpr size_t OFF_XATT  = OFF_Y2 + 8388608;

extern "C" void kernel_launch(void* const* d_in, const int* in_sizes, int n_in,
                              void* d_out, int out_size, void* d_ws, size_t ws_size,
                              hipStream_t stream) {
  (void)in_sizes; (void)n_in; (void)out_size; (void)ws_size;
  const float* x    = (const float*)d_in[0];
  const float* sw   = (const float*)d_in[1];
  const float* scal = (const float*)d_in[2];
  const float* dw   = (const float*)d_in[3];
  const float* b1g  = (const float*)d_in[4];
  const float* b1b  = (const float*)d_in[5];
  const float* b1m  = (const float*)d_in[6];
  const float* b1v  = (const float*)d_in[7];
  const float* pw   = (const float*)d_in[8];
  const float* b2g  = (const float*)d_in[9];
  const float* b2b  = (const float*)d_in[10];
  const float* b2m  = (const float*)d_in[11];
  const float* b2v  = (const float*)d_in[12];
  const float* caw1 = (const float*)d_in[13];
  const float* caw2 = (const float*)d_in[14];
  const float* saw  = (const float*)d_in[15];

  char* ws = (char*)d_ws;
  unsigned short* Mbf  = (unsigned short*)(ws + OFF_MBF);
  unsigned short* Wbf  = (unsigned short*)(ws + OFF_WBF);
  unsigned short* PWbf = (unsigned short*)(ws + OFF_PWBF);
  float* sc1   = (float*)(ws + OFF_SC1);
  float* bi1   = (float*)(ws + OFF_BI1);
  float* sc2   = (float*)(ws + OFF_SC2);
  float* bi2   = (float*)(ws + OFF_BI2);
  float* gapB  = (float*)(ws + OFF_GAP);
  float* cwB   = (float*)(ws + OFF_CW);
  float* mpB   = (float*)(ws + OFF_MP);
  float* sB    = (float*)(ws + OFF_S);
  float* bands = (float*)(ws + OFF_BANDS);
  float* y1    = (float*)(ws + OFF_Y1);
  float* y2    = (float*)(ws + OFF_Y2);
  float* xatt  = (float*)(ws + OFF_XATT);
  float* out   = (float*)d_out;

  // --- prep (bf16 weights, folded BN, band-projection matrices) ---
  prep_mband  <<<64,  256, 0, stream>>>(Mbf);
  prep_wspline<<<512, 256, 0, stream>>>(sw, scal, Wbf);
  prep_pw_bn  <<<16,  256, 0, stream>>>(pw, PWbf, b1g, b1b, b1m, b1v,
                                        b2g, b2b, b2m, b2v, sc1, bi1, sc2, bi2);

  // --- spectral band split as 4x per-band 64x64 GEMM (WMMA) ---
  gemm64_wmma<<<dim3(32, NBB), 256, 0, stream>>>(Mbf, 1, 2, x, bands, nullptr, nullptr);

  // --- depthwise 3x3 + BN1 + ReLU ---
  dwconv_bn_relu<<<8192, 256, 0, stream>>>(bands, dw, sc1, bi1, y1);

  // --- pointwise 1x1 (WMMA) + BN2 ---
  gemm64_wmma<<<dim3(32, NBB), 256, 0, stream>>>(PWbf, 0, 0, y1, y2, sc2, bi2);

  // --- channel attention ---
  gap_kernel<<<512, 256, 0, stream>>>(y2, gapB);
  chan_att  <<<NBB, 64,  0, stream>>>(gapB, caw1, caw2, cwB);

  // --- spatial attention + apply ---
  mean_max_c <<<128,  256, 0, stream>>>(y2, cwB, mpB);
  spatial_att<<<128,  256, 0, stream>>>(mpB, saw, sB);
  apply_att  <<<8192, 256, 0, stream>>>(y2, cwB, sB, xatt);

  // --- cubic B-spline basis + band-fused contraction (WMMA), mean over bands ---
  spline_wmma<<<dim3(32, 2), 256, 0, stream>>>(xatt, Wbf, out);
}
